// UpDownBackbone_16200616641083
// MI455X (gfx1250) — compile-verified, matrix operand order checked by reading, not run on hardware
//
#include <hip/hip_runtime.h>
#include <hip/hip_bf16.h>

// ---------------------------------------------------------------------------
// UpDownBackbone: pos-matching gather + patch edge-energy, CDNA5 (gfx1250)
//
//  B=4, N=4096, C=256, H=W=512, PATCH=4, MIN_PATCH=4, POS_MAX=124
//
//  Kernel 1 (argmin): pos_shuffled is a permutation of pos_org, so
//  min L1 distance == 0 and argmin(L1) == argmin(L2^2). L2^2 argmin over j of
//  (|q_j|^2 - 2 p_i . q_j) is computed exactly with V_WMMA_I32_16X16X64_IU8
//  (coords < 124 fit u8; products/accums exact in i32). 2x-unrolled with
//  independent B/acc registers to hide the IU8 WMMA->VALU hazard slots.
// ---------------------------------------------------------------------------

typedef __attribute__((ext_vector_type(8))) int v8i;

#define BB   4
#define NN   4096
#define CC   256
#define HH   512
#define WW   512

__global__ __launch_bounds__(256)
void argmin_wmma_kernel(const int* __restrict__ pos_org,
                        const int* __restrict__ pos_shuf,
                        int* __restrict__ idx_out)
{
    // 8 waves / block; 32 blocks per batch -> 256 i-tiles per batch
    __shared__ unsigned int spos[NN];   // packed x | y<<8          (16 KB)
    __shared__ unsigned int sqnb[NN];   // |q|^2 + bias (key high)  (16 KB)

    const int lane = threadIdx.x & 31;
    const int wave = threadIdx.x >> 5;
    const int b    = blockIdx.x >> 5;                  // 32 blocks per batch
    const int it   = ((blockIdx.x & 31) << 3) + wave;  // i-tile 0..255
    const int col  = lane & 15;                        // column within tile

    // Stage candidate positions (packed u8 coords) + biased norms into LDS
    for (int j = threadIdx.x; j < NN; j += 256) {
        const int x = pos_shuf[((size_t)b * NN + j) * 2 + 0];
        const int y = pos_shuf[((size_t)b * NN + j) * 2 + 1];
        spos[j] = (unsigned)x | ((unsigned)y << 8);
        sqnb[j] = (unsigned)(x * x + y * y + (1 << 18));
    }
    __syncthreads();

    // A-matrix (16x64 u8): lanes 0-15 hold row M=lane, VGPR0 bytes = K0..K3.
    // Only K0=x, K1=y nonzero; lanes 16-31 hold K8..15 == 0.
    const int row = it * 16 + col;
    const int xa = pos_org[((size_t)b * NN + row) * 2 + 0];
    const int ya = pos_org[((size_t)b * NN + row) * 2 + 1];
    v8i amat = {};
    amat[0] = (lane < 16) ? (xa | (ya << 8)) : 0;

    // B operands: zero-filled once; only element 0 is rewritten per tile.
    // Two independent vectors so back-to-back WMMAs have no WAR hazard.
    v8i bmat0 = {};
    v8i bmat1 = {};

    // Running best key per covered row: (score+bias)<<32 | j  (unsigned min
    // preserves "first index of minimum" tie-breaking of jnp.argmin).
    unsigned long long best[8];
#pragma unroll
    for (int r = 0; r < 8; ++r) best[r] = ~0ull;

    for (int jt = 0; jt < NN / 16; jt += 2) {
        const unsigned p0 = spos[jt * 16 + col];
        const unsigned p1 = spos[jt * 16 + 16 + col];
        const unsigned q0 = sqnb[jt * 16 + col];        // |q|^2 + bias
        const unsigned q1 = sqnb[jt * 16 + 16 + col];

        // B-matrix (64x16 u8): lanes 0-15 VGPR0 = K0..K3 of column `lane`;
        // lanes 16-31 VGPR0 = K16..19 == 0.
        bmat0[0] = (lane < 16) ? (int)p0 : 0;
        bmat1[0] = (lane < 16) ? (int)p1 : 0;

        v8i cz = {};
        v8i acc0 = __builtin_amdgcn_wmma_i32_16x16x64_iu8(
            false, amat, false, bmat0, cz, false, false);
        v8i acc1 = __builtin_amdgcn_wmma_i32_16x16x64_iu8(
            false, amat, false, bmat1, cz, false, false);

        const unsigned jg0 = (unsigned)(jt * 16 + col);
        const unsigned jg1 = jg0 + 16u;
#pragma unroll
        for (int r = 0; r < 8; ++r) {
            // row M = r (lanes 0-15) or 8+r (lanes 16-31); constant |p_i|^2
            // dropped (does not affect argmin over j)
            const unsigned s0 = q0 - (unsigned)(2 * acc0[r]);
            const unsigned long long k0 = ((unsigned long long)s0 << 32) | jg0;
            best[r] = k0 < best[r] ? k0 : best[r];
        }
#pragma unroll
        for (int r = 0; r < 8; ++r) {
            const unsigned s1 = q1 - (unsigned)(2 * acc1[r]);
            const unsigned long long k1 = ((unsigned long long)s1 << 32) | jg1;
            best[r] = k1 < best[r] ? k1 : best[r];
        }
    }

    // Cross-lane min within each 16-lane half (masks <16 stay in-half, wave32)
#pragma unroll
    for (int r = 0; r < 8; ++r) {
        unsigned long long k = best[r];
#pragma unroll
        for (int m = 1; m <= 8; m <<= 1) {
            const unsigned long long o = __shfl_xor(k, m, 32);
            k = o < k ? o : k;
        }
        best[r] = k;
    }

    // lanes 0-15 carry rows M=0..7 (VGPR r -> M=r); lanes 16-31 rows 8..15
    if ((lane & 15) == 0) {
        const int half = lane >> 4;                    // 0 or 1
#pragma unroll
        for (int r = 0; r < 8; ++r)
            idx_out[(size_t)b * NN + it * 16 + half * 8 + r] =
                (int)(best[r] & 0xFFFFFFFFu);
    }
}

// ---------------------------------------------------------------------------
// Kernel 2: feat_r[b,i,:] = feat[b, idx[b,i], :]   (float4 vectorized)
// ---------------------------------------------------------------------------
__global__ __launch_bounds__(256)
void gather_feat_kernel(const float* __restrict__ feat,
                        const int* __restrict__ idx,
                        float* __restrict__ out)
{
    const int t  = blockIdx.x * blockDim.x + threadIdx.x;   // B*N*64 threads
    const int c4 = t & 63;                                  // 64 float4 per row
    const int i  = (t >> 6) & (NN - 1);
    const int b  = t >> 18;
    const int j  = idx[(size_t)b * NN + i];
    const float4* src = (const float4*)(feat + ((size_t)b * NN + j) * CC);
    float4*       dst = (float4*)(out  + ((size_t)b * NN + i) * CC);
    __builtin_prefetch(&src[c4], 0, 1);                     // global_prefetch_b8
    dst[c4] = src[c4];
}

// ---------------------------------------------------------------------------
// Kernel 3: dis[b,i] = sum of on-the-fly edge map over the 4x4 block at
//           (row0 = pos_y*4, col0 = pos_x*4)
// ---------------------------------------------------------------------------
__device__ __forceinline__ float color_dist(const float* __restrict__ im,
                                            int o0, int o1)
{
    const int P = HH * WW;
    return fabsf(im[o0]         - im[o1])
         + fabsf(im[P + o0]     - im[P + o1])
         + fabsf(im[2 * P + o0] - im[2 * P + o1]);
}

__device__ __forceinline__ float edge_at(const float* __restrict__ im,
                                         int y, int x)
{
    const int o = y * WW + x;
    float e = 0.0f;
    if (y >= 1)      e += color_dist(im, o, o - WW);   // cd(y, y-1)
    if (y <= HH - 2) e += color_dist(im, o + WW, o);   // cd(y+1, y)
    if (x >= 1)      e += color_dist(im, o, o - 1);    // cd(x, x-1)
    if (x <= WW - 2) e += color_dist(im, o + 1, o);    // cd(x+1, x)
    return e;
}

__global__ __launch_bounds__(256)
void patch_dis_kernel(const float* __restrict__ images,
                      const int* __restrict__ pos_org,
                      float* __restrict__ dis)
{
    const int t = blockIdx.x * blockDim.x + threadIdx.x;    // B*N threads
    if (t >= BB * NN) return;
    const int i = t & (NN - 1);
    const int b = t >> 12;
    const int px = pos_org[((size_t)b * NN + i) * 2 + 0] * 4;  // col base (x)
    const int py = pos_org[((size_t)b * NN + i) * 2 + 1] * 4;  // row base (y)
    const float* im = images + (size_t)b * 3 * HH * WW;
    float s = 0.0f;
#pragma unroll
    for (int dy = 0; dy < 4; ++dy)
#pragma unroll
        for (int dx = 0; dx < 4; ++dx)
            s += edge_at(im, py + dy, px + dx);
    dis[t] = s;
}

// ---------------------------------------------------------------------------
extern "C" void kernel_launch(void* const* d_in, const int* in_sizes, int n_in,
                              void* d_out, int out_size, void* d_ws, size_t ws_size,
                              hipStream_t stream)
{
    (void)in_sizes; (void)n_in; (void)out_size; (void)ws_size;

    const float* feat     = (const float*)d_in[0];   // (4,4096,256) f32
    const float* images   = (const float*)d_in[1];   // (4,3,512,512) f32
    const int*   pos_org  = (const int*)d_in[2];     // (4,4096,2) i32
    const int*   pos_shuf = (const int*)d_in[3];     // (4,4096,2) i32
    // d_in[4]=patch_size(4), d_in[5]=min_patch_size(4): compile-time constants

    float* out_feat = (float*)d_out;                        // 4*4096*256 floats
    float* out_dis  = (float*)d_out + (size_t)BB * NN * CC; // + 4*4096 floats
    int*   idx      = (int*)d_ws;                           // 64 KB scratch

    // 1) WMMA IU8 argmin: 128 blocks x 8 waves = 1024 i-tiles (4 batches x 256)
    argmin_wmma_kernel<<<128, 256, 0, stream>>>(pos_org, pos_shuf, idx);

    // 2) Gather: B*N*64 float4 threads
    gather_feat_kernel<<<(BB * NN * 64) / 256, 256, 0, stream>>>(feat, idx, out_feat);

    // 3) Patch edge-energy sums: B*N threads
    patch_dis_kernel<<<(BB * NN) / 256, 256, 0, stream>>>(images, pos_org, out_dis);
}
